// AlbertLayer_16011638079995
// MI455X (gfx1250) — matrix operand order repaired and checked by reading, not compiled, not run on hardware
//
#include <hip/hip_runtime.h>

typedef __bf16 bf16;
typedef __attribute__((ext_vector_type(16))) __bf16 v16bf;
typedef __attribute__((ext_vector_type(8)))  __bf16 v8bf;
typedef __attribute__((ext_vector_type(8)))  float  v8f;
typedef __attribute__((ext_vector_type(4)))  unsigned int v4u;
typedef __attribute__((ext_vector_type(4)))  int v4i;
typedef __attribute__((ext_vector_type(8)))  int v8i;

constexpr int cB = 2, cS = 2048, cH = 1024, cNH = 16, cHD = 64, cI = 4096, cE = 8;
constexpr int cT = cB * cS;  // 4096 tokens

#define WMMA_BF16(a, b, c) \
  __builtin_amdgcn_wmma_f32_16x16x32_bf16(false, (a), false, (b), (short)0, (c), false, false)

#if __has_builtin(__builtin_amdgcn_tensor_load_to_lds)
#define HAVE_TDM 1
#else
#define HAVE_TDM 0
#endif

static __device__ __forceinline__ bf16 f2bf(float f) {
  unsigned int u = __float_as_uint(f);
  u += 0x7fffu + ((u >> 16) & 1u);             // round-to-nearest-even
  unsigned short h = (unsigned short)(u >> 16);
  return __builtin_bit_cast(bf16, h);
}

#if HAVE_TDM
// Issue one TDM 2D tile load: 32 rows x 128 cols of bf16 from row-major global
// (row stride ldb elements) into LDS at lds_addr, padded +1 DWORD per row
// (64-DWORD interval) => LDS row stride 130 bf16 elements. TENSORcnt-tracked.
static __device__ __forceinline__ void tdm_issue_b(const bf16* src, int ldb, int Ktot,
                                                   unsigned lds_addr) {
  unsigned long long ga = (unsigned long long)(const void*)src;
  v4u g0;
  g0[0] = 1u;                                            // count=1, user mode
  g0[1] = lds_addr;                                      // LDS byte address
  g0[2] = (unsigned)(ga & 0xffffffffu);                  // global_addr[31:0]
  g0[3] = (unsigned)((ga >> 32) & 0x01ffffffu) | (2u << 30);  // addr[56:32] | type=2
  v8i g1;
  // data_size=1 (2B) | pad_enable | pad_interval=5 (64 DW) | pad_amount=0 (1 DW)
  g1[0] = (int)((1u << 16) | (1u << 20) | (5u << 22));
  g1[1] = (int)(((unsigned)ldb & 0xffffu) << 16);        // tensor_dim0[15:0] @ [63:48]
  g1[2] = (int)((((unsigned)ldb >> 16) & 0xffffu) |      // tensor_dim0[31:16]
                (((unsigned)Ktot & 0xffffu) << 16));     // tensor_dim1[15:0]
  g1[3] = (int)((((unsigned)Ktot >> 16) & 0xffffu) |     // tensor_dim1[31:16]
                (128u << 16));                           // tile_dim0 = 128
  g1[4] = 32;                                            // tile_dim1 = 32
  g1[5] = (int)(unsigned)ldb;                            // tensor_dim0_stride[31:0]
  g1[6] = 0;
  g1[7] = 0;
  v4i z4 = {};
#if __clang_major__ >= 23
  v8i z8 = {};
  __builtin_amdgcn_tensor_load_to_lds(g0, g1, z4, z4, z8, 0);
#else
  __builtin_amdgcn_tensor_load_to_lds(g0, g1, z4, z4, 0);
#endif
}
#endif

// ---------------------------------------------------------------------------
// fp32 -> bf16 cast (grid-strided)
// ---------------------------------------------------------------------------
__global__ __launch_bounds__(256) void cast_bf16_kernel(const float* __restrict__ src,
                                                        bf16* __restrict__ dst, long n) {
  long i = (long)blockIdx.x * blockDim.x + threadIdx.x;
  long stride = (long)gridDim.x * blockDim.x;
  for (; i < n; i += stride) dst[i] = f2bf(src[i]);
}

// ---------------------------------------------------------------------------
// Generic bf16 WMMA GEMM: C[M,N] = A[M,K] x B[K,N]
// Block tile 64x128, BK=32, 8 waves: wave (wm,wn) owns a 16x64 strip (4 WMMAs).
// B tiles arrive via TDM (double-buffered, TENSORcnt) when available.
// MODE 0: C fp32 = acc
// MODE 1: C fp32 = acc + bias[n]
// MODE 2: A rows gathered via rowidx; C bf16 = gelu(acc) at LOCAL rows
// MODE 3: A local rows; C fp32 [rowidx[row]] += rowscale[row] * acc
// ---------------------------------------------------------------------------
template <int MODE>
__global__ __launch_bounds__(256) void gemm_bf16_kernel(
    const bf16* __restrict__ A, int lda, const bf16* __restrict__ Bm, int ldb,
    int M, int N, int K,
    float* __restrict__ Cf, bf16* __restrict__ Cb, const float* __restrict__ bias,
    const int* __restrict__ rowidx, const float* __restrict__ rowscale,
    const int* __restrict__ count) {
  constexpr int AS_LD = 40;   // 80B rows: 16B-aligned, bank-rotated (gcd(20,64)=4)
  constexpr int BS_LD = 130;  // matches TDM pad: 64 DW + 1 DW pad per row
  __shared__ __align__(16) bf16 As[64 * AS_LD];
  __shared__ __align__(16) bf16 Bs[2][32 * BS_LD];

  const int tid = threadIdx.x;
  const int lane = tid & 31;
  const int wave = tid >> 5;
  const int wm = wave >> 1;  // 0..3
  const int wn = wave & 1;   // 0..1
  const int m0 = blockIdx.y * 64;
  const int n0 = blockIdx.x * 128;

  int mlive = M;
  if (count != nullptr) {
    mlive = *count;
    if (m0 >= mlive) return;
  }

  v8f acc[4] = {};
  const int lm = lane & 15, lh = lane >> 4;
  const int nk = K >> 5;

#if HAVE_TDM
  const unsigned bladdr0 = (unsigned)(unsigned long long)(const void*)&Bs[0][0];
  const unsigned bladdr1 = (unsigned)(unsigned long long)(const void*)&Bs[1][0];
  if (wave == 0) tdm_issue_b(Bm + n0, ldb, K, bladdr0);
#endif

  for (int t = 0; t < nk; ++t) {
    const int k0 = t << 5;
    // ---- stage A tile: 64 rows x 32 cols, one 16B load/store per thread ----
    {
      const int r = tid >> 2;
      const int c = (tid & 3) * 8;
      const int gm = m0 + r;
      long grow;
      if constexpr (MODE == 2) {
        grow = (gm < mlive) ? (long)rowidx[gm] : -1;
      } else {
        grow = (count != nullptr && gm >= mlive) ? -1 : (long)gm;
      }
      if (grow >= 0) {
        *(v8bf*)(As + r * AS_LD + c) = *(const v8bf*)(A + grow * (long)lda + k0 + c);
      } else {
        v8bf z = {};
        *(v8bf*)(As + r * AS_LD + c) = z;
      }
    }
#if HAVE_TDM
    // ---- async B tile pipeline: issue next, wait for current ----
    if (wave == 0) {
      if (t + 1 < nk) {
        tdm_issue_b(Bm + (long)(k0 + 32) * ldb + n0, ldb, K,
                    ((t + 1) & 1) ? bladdr1 : bladdr0);
        __builtin_amdgcn_s_wait_tensorcnt((short)1);
      } else {
        __builtin_amdgcn_s_wait_tensorcnt((short)0);
      }
    }
#else
    // ---- manual B staging: 32 rows x 128 cols, 16 bf16 per thread ----
    {
      const int r = tid >> 3;
      const int c = (tid & 7) * 16;
      const bf16* p = Bm + (long)(k0 + r) * ldb + n0 + c;
      bf16* q = &Bs[t & 1][r * BS_LD + c];
#pragma unroll
      for (int e = 0; e < 16; ++e) q[e] = p[e];
      if (k0 + 32 < K)
        __builtin_prefetch((const void*)(Bm + (long)(k0 + 32 + r) * ldb + n0 + c), 0, 1);
    }
#endif
    __syncthreads();

    // ---- A fragment: two aligned 16B LDS reads (ISA 16-bit A layout) ----
    v16bf afrag;
    {
      const v8bf* ar = (const v8bf*)(As + (wm * 16 + lm) * AS_LD);
      v8bf c0 = ar[lh];       // K = 8h .. 8h+7
      v8bf c1 = ar[2 + lh];   // K = 16+8h .. 16+8h+7
      afrag = __builtin_shufflevector(c0, c1, 0, 1, 2, 3, 4, 5, 6, 7,
                                      8, 9, 10, 11, 12, 13, 14, 15);
    }
    // ---- 4 B fragments + WMMA ----
    const bf16* Bsb = Bs[t & 1];
#pragma unroll
    for (int f = 0; f < 4; ++f) {
      v16bf bfrag;
      const int n = wn * 64 + f * 16 + lm;
#pragma unroll
      for (int e = 0; e < 16; ++e) bfrag[e] = Bsb[(8 * lh + e + (e >= 8 ? 8 : 0)) * BS_LD + n];
      acc[f] = WMMA_BF16(afrag, bfrag, acc[f]);
    }
    __syncthreads();
  }

  // ---- epilogue (C layout: M = r + 8*lh, N = lm) ----
#pragma unroll
  for (int f = 0; f < 4; ++f) {
#pragma unroll
    for (int r = 0; r < 8; ++r) {
      const int gm = m0 + wm * 16 + r + 8 * lh;
      const int gn = n0 + wn * 64 + f * 16 + lm;
      if (count != nullptr && gm >= mlive) continue;
      const float v = acc[f][r];
      if constexpr (MODE == 0) {
        Cf[(long)gm * N + gn] = v;
      } else if constexpr (MODE == 1) {
        Cf[(long)gm * N + gn] = v + bias[gn];
      } else if constexpr (MODE == 2) {
        const float g = 0.5f * v * (1.0f + erff(v * 0.70710678118f));
        Cb[(long)gm * N + gn] = f2bf(g);
      } else {
        const int orow = rowidx[gm];
        Cf[(long)orow * N + gn] += rowscale[gm] * v;
      }
    }
  }
}

// ---------------------------------------------------------------------------
// RoPE + head-pack: fp32 [B,S,H] -> bf16 [B,NH,S,HD] for Q,K (rotated) and V
// ---------------------------------------------------------------------------
__global__ __launch_bounds__(256) void rope_pack_kernel(
    const float* __restrict__ qf, const float* __restrict__ kf, const float* __restrict__ vf,
    bf16* __restrict__ Qp, bf16* __restrict__ Kp, bf16* __restrict__ Vp) {
  const long i = (long)blockIdx.x * 256 + threadIdx.x;  // B*NH*S*32 threads
  const int j = (int)(i & 31);
  long r = i >> 5;
  const int s = (int)(r % cS);
  r /= cS;
  const int nh = (int)(r % cNH);
  const int b = (int)(r / cNH);

  const long src = ((long)(b * cS + s)) * cH + nh * cHD;
  const long dst = ((long)((b * cNH + nh)) * cS + s) * cHD;

  // inv_freq = 10000^(-2j/HD) = exp(-ln(10000) * j / 32)
  const float ang = (float)s * __expf(-9.210340371976184f * (float)j / 32.0f);
  const float c = __cosf(ang), sn = __sinf(ang);

  const float q1 = qf[src + j], q2 = qf[src + j + 32];
  Qp[dst + j] = f2bf(q1 * c - q2 * sn);
  Qp[dst + j + 32] = f2bf(q2 * c + q1 * sn);

  const float k1 = kf[src + j], k2 = kf[src + j + 32];
  Kp[dst + j] = f2bf(k1 * c - k2 * sn);
  Kp[dst + j + 32] = f2bf(k2 * c + k1 * sn);

  Vp[dst + j] = f2bf(vf[src + j]);
  Vp[dst + j + 32] = f2bf(vf[src + j + 32]);
}

// ---------------------------------------------------------------------------
// Flash attention: grid (S/128, NH, B), 8 waves; each wave owns 16 Q rows.
// Scores & PV via WMMA; online softmax; ctx written bf16 [B,S,H] (= [T,H]).
// ---------------------------------------------------------------------------
__global__ __launch_bounds__(256) void flash_attn_kernel(
    const bf16* __restrict__ Qp, const bf16* __restrict__ Kp, const bf16* __restrict__ Vp,
    bf16* __restrict__ ctxb) {
  __shared__ float Pt[8][16 * 33];  // per-wave 16x32 P transpose buffer

  const int lane = threadIdx.x & 31;
  const int wave = threadIdx.x >> 5;
  const int lm = lane & 15, lh = lane >> 4;
  const int b = blockIdx.z, nh = blockIdx.y;
  const int q0 = blockIdx.x * 128 + wave * 16;
  const long base = ((long)(b * cNH + nh)) * cS * cHD;

  // Q fragments (hd 0..31 and 32..63)
  v16bf aq0, aq1;
  {
    const bf16* qrow = Qp + base + (long)(q0 + lm) * cHD;
#pragma unroll
    for (int e = 0; e < 8; ++e) {
      aq0[e] = qrow[8 * lh + e];
      aq0[8 + e] = qrow[16 + 8 * lh + e];
      aq1[e] = qrow[32 + 8 * lh + e];
      aq1[8 + e] = qrow[48 + 8 * lh + e];
    }
  }

  v8f o[4] = {};
  float mrow[8], lrow[8];
#pragma unroll
  for (int r = 0; r < 8; ++r) { mrow[r] = -3.0e38f; lrow[r] = 0.0f; }

  for (int kc = 0; kc < cS; kc += 32) {
    // ---- scores for 32 keys: two 16x16 tiles ----
    v8f s0 = {}, s1 = {};
    {
      const bf16* kr0 = Kp + base + (long)(kc + lm) * cHD;
      const bf16* kr1 = kr0 + 16 * cHD;
      v16bf bk;
#pragma unroll
      for (int e = 0; e < 16; ++e) bk[e] = kr0[8 * lh + e + (e >= 8 ? 8 : 0)];
      s0 = WMMA_BF16(aq0, bk, s0);
#pragma unroll
      for (int e = 0; e < 16; ++e) bk[e] = kr0[32 + 8 * lh + e + (e >= 8 ? 8 : 0)];
      s0 = WMMA_BF16(aq1, bk, s0);
#pragma unroll
      for (int e = 0; e < 16; ++e) bk[e] = kr1[8 * lh + e + (e >= 8 ? 8 : 0)];
      s1 = WMMA_BF16(aq0, bk, s1);
#pragma unroll
      for (int e = 0; e < 16; ++e) bk[e] = kr1[32 + 8 * lh + e + (e >= 8 ? 8 : 0)];
      s1 = WMMA_BF16(aq1, bk, s1);
    }
    // ---- online softmax, P -> LDS ----
#pragma unroll
    for (int r = 0; r < 8; ++r) {
      const float v0 = s0[r] * 0.125f;  // 1/sqrt(64)
      const float v1 = s1[r] * 0.125f;
      float mx = fmaxf(v0, v1);
#pragma unroll
      for (int off = 1; off < 16; off <<= 1) mx = fmaxf(mx, __shfl_xor(mx, off, 32));
      const float nm = fmaxf(mrow[r], mx);
      const float corr = __expf(mrow[r] - nm);
      const float p0 = __expf(v0 - nm);
      const float p1 = __expf(v1 - nm);
      float rs = p0 + p1;
#pragma unroll
      for (int off = 1; off < 16; off <<= 1) rs += __shfl_xor(rs, off, 32);
      lrow[r] = lrow[r] * corr + rs;
      mrow[r] = nm;
      o[0][r] *= corr; o[1][r] *= corr; o[2][r] *= corr; o[3][r] *= corr;
      const int row = r + 8 * lh;
      Pt[wave][row * 33 + lm] = p0;
      Pt[wave][row * 33 + 16 + lm] = p1;
    }
    __syncthreads();
    // ---- P as A fragment (K = key offset 0..31) ----
    v16bf pa;
#pragma unroll
    for (int e = 0; e < 8; ++e) {
      pa[e] = f2bf(Pt[wave][lm * 33 + 8 * lh + e]);
      pa[8 + e] = f2bf(Pt[wave][lm * 33 + 16 + 8 * lh + e]);
    }
    // ---- P x V ----
#pragma unroll
    for (int f = 0; f < 4; ++f) {
      v16bf vb;
      const int hd = f * 16 + lm;
      const bf16* vr = Vp + base + (long)(kc + 8 * lh) * cHD + hd;
#pragma unroll
      for (int e = 0; e < 16; ++e) vb[e] = vr[(long)(e + (e >= 8 ? 8 : 0)) * cHD];
      o[f] = WMMA_BF16(pa, vb, o[f]);
    }
    __syncthreads();
  }

  // ---- normalize + store ctx (bf16, [T,H] layout) ----
#pragma unroll
  for (int r = 0; r < 8; ++r) {
    const float inv = 1.0f / lrow[r];
    const int q = q0 + r + 8 * lh;
#pragma unroll
    for (int f = 0; f < 4; ++f) {
      const int col = nh * cHD + f * 16 + lm;
      ctxb[((long)(b * cS + q)) * cH + col] = f2bf(o[f][r] * inv);
    }
  }
}

// ---------------------------------------------------------------------------
// LayerNorm over H=1024: out = LN(a + res) * g + b ; optional bf16 copy
// ---------------------------------------------------------------------------
__global__ __launch_bounds__(256) void layernorm_kernel(
    const float* __restrict__ a, const float* __restrict__ res,
    const float* __restrict__ g, const float* __restrict__ bta,
    float* __restrict__ outf, bf16* __restrict__ outb) {
  __shared__ float rs1[8], rs2[8];
  const long t = blockIdx.x;
  const int lane = threadIdx.x & 31, wv = threadIdx.x >> 5;

  float vloc[4], s = 0.0f, s2 = 0.0f;
#pragma unroll
  for (int j = 0; j < 4; ++j) {
    const int i = threadIdx.x + j * 256;
    float v = a[t * cH + i];
    if (res != nullptr) v += res[t * cH + i];
    vloc[j] = v;
    s += v;
    s2 += v * v;
  }
#pragma unroll
  for (int off = 16; off > 0; off >>= 1) {
    s += __shfl_xor(s, off, 32);
    s2 += __shfl_xor(s2, off, 32);
  }
  if (lane == 0) { rs1[wv] = s; rs2[wv] = s2; }
  __syncthreads();
  float S1 = 0.0f, S2 = 0.0f;
#pragma unroll
  for (int w = 0; w < 8; ++w) { S1 += rs1[w]; S2 += rs2[w]; }
  const float mu = S1 / (float)cH;
  const float var = S2 / (float)cH - mu * mu;
  const float rinv = rsqrtf(var + 1e-12f);
#pragma unroll
  for (int j = 0; j < 4; ++j) {
    const int i = threadIdx.x + j * 256;
    const float ov = (vloc[j] - mu) * rinv * g[i] + bta[i];
    outf[t * cH + i] = ov;
    if (outb != nullptr) outb[t * cH + i] = f2bf(ov);
  }
}

// ---------------------------------------------------------------------------
// Router: softmax over E=8 logits, top-2, build per-expert token lists.
// ---------------------------------------------------------------------------
__global__ __launch_bounds__(256) void router_kernel(
    const float* __restrict__ hs, const float* __restrict__ Wg,
    float* __restrict__ Psum, int* __restrict__ counts,
    int* __restrict__ idx, float* __restrict__ scales) {
  __shared__ float wg[cH * cE];  // 32 KB
  for (int i = threadIdx.x; i < cH * cE; i += 256) wg[i] = Wg[i];
  __syncthreads();

  const long t = (long)blockIdx.x * 256 + threadIdx.x;
  float acc[cE] = {};
  const float* x = hs + t * cH;
  for (int i = 0; i < cH; ++i) {
    const float xv = x[i];
#pragma unroll
    for (int e = 0; e < cE; ++e) acc[e] += xv * wg[i * cE + e];
  }
  float mx = acc[0];
#pragma unroll
  for (int e = 1; e < cE; ++e) mx = fmaxf(mx, acc[e]);
  float p[cE], sum = 0.0f;
#pragma unroll
  for (int e = 0; e < cE; ++e) { p[e] = __expf(acc[e] - mx); sum += p[e]; }
#pragma unroll
  for (int e = 0; e < cE; ++e) p[e] /= sum;

  int e0 = 0;
#pragma unroll
  for (int e = 1; e < cE; ++e) if (p[e] > p[e0]) e0 = e;
  int e1 = (e0 == 0) ? 1 : 0;
#pragma unroll
  for (int e = 0; e < cE; ++e) if (e != e0 && p[e] > p[e1]) e1 = e;

  const float w0 = p[e0], w1 = p[e1], ws = w0 + w1;
  const int pos0 = atomicAdd(&counts[e0], 1);
  idx[(long)e0 * cT + pos0] = (int)t;
  scales[(long)e0 * cT + pos0] = w0 / ws;
  const int pos1 = atomicAdd(&counts[e1], 1);
  idx[(long)e1 * cT + pos1] = (int)t;
  scales[(long)e1 * cT + pos1] = w1 / ws;
#pragma unroll
  for (int e = 0; e < cE; ++e) atomicAdd(&Psum[e], p[e]);
}

__global__ void aux_loss_kernel(const int* __restrict__ counts,
                                const float* __restrict__ Psum,
                                float* __restrict__ out_aux) {
  if (threadIdx.x == 0 && blockIdx.x == 0) {
    float a = 0.0f;
    for (int e = 0; e < cE; ++e)
      a += ((float)counts[e] / (float)cT) * (Psum[e] / (float)cT);
    *out_aux = (float)cE * a;
  }
}

// ---------------------------------------------------------------------------
// Host orchestration
// ---------------------------------------------------------------------------
extern "C" void kernel_launch(void* const* d_in, const int* in_sizes, int n_in,
                              void* d_out, int out_size, void* d_ws, size_t ws_size,
                              hipStream_t stream) {
  (void)in_sizes; (void)n_in; (void)out_size; (void)ws_size;
  const float* x    = (const float*)d_in[0];
  const float* Wq   = (const float*)d_in[1];
  const float* Wk   = (const float*)d_in[2];
  const float* Wv   = (const float*)d_in[3];
  const float* Wd   = (const float*)d_in[4];
  const float* bd   = (const float*)d_in[5];
  const float* ln1g = (const float*)d_in[6];
  const float* ln1b = (const float*)d_in[7];
  const float* Wg   = (const float*)d_in[8];
  const float* W1   = (const float*)d_in[9];
  const float* W2   = (const float*)d_in[10];
  const float* ln2g = (const float*)d_in[11];
  const float* ln2b = (const float*)d_in[12];
  float* out = (float*)d_out;

  char* base = (char*)d_ws;
  size_t off = 0;
  auto alloc = [&](size_t bytes) -> void* {
    void* p = base + off;
    off += (bytes + 255) & ~(size_t)255;
    return p;
  };
  const size_t TH = (size_t)cT * cH;
  const size_t TI = (size_t)cT * cI;
  const size_t W12 = (size_t)cE * cH * cI;

  bf16* xb   = (bf16*)alloc(TH * 2);
  bf16* Wqb  = (bf16*)alloc((size_t)cH * cH * 2);
  bf16* Wkb  = (bf16*)alloc((size_t)cH * cH * 2);
  bf16* Wvb  = (bf16*)alloc((size_t)cH * cH * 2);
  bf16* Wdb  = (bf16*)alloc((size_t)cH * cH * 2);
  bf16* W1b  = (bf16*)alloc(W12 * 2);
  bf16* W2b  = (bf16*)alloc(W12 * 2);
  float* qf  = (float*)alloc(TH * 4);
  float* kf  = (float*)alloc(TH * 4);
  float* vf  = (float*)alloc(TH * 4);
  bf16* Qp   = (bf16*)alloc(TH * 2);
  bf16* Kp   = (bf16*)alloc(TH * 2);
  bf16* Vp   = (bf16*)alloc(TH * 2);
  bf16* ctxb = (bf16*)alloc(TH * 2);
  float* attn= (float*)alloc(TH * 4);
  float* hs  = (float*)alloc(TH * 4);
  bf16* hsb  = (bf16*)alloc(TH * 2);
  float* moe = (float*)alloc(TH * 4);
  bf16* h1b  = (bf16*)alloc(TI * 2);
  int* counts= (int*)alloc(cE * 4);
  float* Psum= (float*)alloc(cE * 4);
  int* idx   = (int*)alloc((size_t)cE * cT * 4);
  float* scl = (float*)alloc((size_t)cE * cT * 4);

  // per-call zeroing (graph-capturable)
  hipMemsetAsync(counts, 0, cE * 4, stream);
  hipMemsetAsync(Psum, 0, cE * 4, stream);
  hipMemsetAsync(moe, 0, TH * 4, stream);

  // fp32 -> bf16 staging
  cast_bf16_kernel<<<2048, 256, 0, stream>>>(x, xb, (long)TH);
  cast_bf16_kernel<<<1024, 256, 0, stream>>>(Wq, Wqb, (long)cH * cH);
  cast_bf16_kernel<<<1024, 256, 0, stream>>>(Wk, Wkb, (long)cH * cH);
  cast_bf16_kernel<<<1024, 256, 0, stream>>>(Wv, Wvb, (long)cH * cH);
  cast_bf16_kernel<<<1024, 256, 0, stream>>>(Wd, Wdb, (long)cH * cH);
  cast_bf16_kernel<<<8192, 256, 0, stream>>>(W1, W1b, (long)W12);
  cast_bf16_kernel<<<8192, 256, 0, stream>>>(W2, W2b, (long)W12);

  // QKV projections (WMMA GEMM, TDM-fed B tiles)
  dim3 gqkv(cH / 128, cT / 64);
  gemm_bf16_kernel<0><<<gqkv, 256, 0, stream>>>(xb, cH, Wqb, cH, cT, cH, cH,
                                                qf, nullptr, nullptr, nullptr, nullptr, nullptr);
  gemm_bf16_kernel<0><<<gqkv, 256, 0, stream>>>(xb, cH, Wkb, cH, cT, cH, cH,
                                                kf, nullptr, nullptr, nullptr, nullptr, nullptr);
  gemm_bf16_kernel<0><<<gqkv, 256, 0, stream>>>(xb, cH, Wvb, cH, cT, cH, cH,
                                                vf, nullptr, nullptr, nullptr, nullptr, nullptr);

  // RoPE + pack to [B,NH,S,HD] bf16
  rope_pack_kernel<<<(cB * cNH * cS * 32) / 256, 256, 0, stream>>>(qf, kf, vf, Qp, Kp, Vp);

  // Flash attention -> ctx bf16 [T,H]
  flash_attn_kernel<<<dim3(cS / 128, cNH, cB), 256, 0, stream>>>(Qp, Kp, Vp, ctxb);

  // Output projection + bias
  gemm_bf16_kernel<1><<<gqkv, 256, 0, stream>>>(ctxb, cH, Wdb, cH, cT, cH, cH,
                                                attn, nullptr, bd, nullptr, nullptr, nullptr);

  // LN1 over (attn + x)
  layernorm_kernel<<<cT, 256, 0, stream>>>(attn, x, ln1g, ln1b, hs, hsb);

  // Router (top-2 of 8) + aux-loss stats
  router_kernel<<<cT / 256, 256, 0, stream>>>(hs, Wg, Psum, counts, idx, scl);

  // Routed MoE: per-expert grouped GEMMs (count-guarded)
  for (int e = 0; e < cE; ++e) {
    gemm_bf16_kernel<2><<<dim3(cI / 128, cT / 64), 256, 0, stream>>>(
        hsb, cH, W1b + (size_t)e * cH * cI, cI, cT, cI, cH,
        nullptr, h1b, nullptr, idx + (size_t)e * cT, nullptr, counts + e);
    gemm_bf16_kernel<3><<<dim3(cH / 128, cT / 64), 256, 0, stream>>>(
        h1b, cI, W2b + (size_t)e * cI * cH, cH, cT, cH, cI,
        moe, nullptr, nullptr, idx + (size_t)e * cT, scl + (size_t)e * cT, counts + e);
  }

  // LN2 over (hs + moe) -> output
  layernorm_kernel<<<cT, 256, 0, stream>>>(hs, moe, ln2g, ln2b, out, nullptr);

  // aux loss scalar -> d_out[T*H]
  aux_loss_kernel<<<1, 1, 0, stream>>>(counts, Psum, out + TH);
}
